// ChannelSelfAttention_50259707298473
// MI455X (gfx1250) — matrix operand order repaired and checked by reading, not compiled, not run on hardware
//
#include <hip/hip_runtime.h>
#include <hip/hip_bf16.h>
#include <stdint.h>

typedef __bf16 bf16_t;
typedef __attribute__((ext_vector_type(16))) __bf16 v16bf;
typedef __attribute__((ext_vector_type(8)))  __bf16 v8bf;
typedef __attribute__((ext_vector_type(8)))  float  v8f;
typedef __attribute__((ext_vector_type(4)))  float  v4f;

// Async DMA: global -> LDS, 16 bytes per lane, tracked by ASYNCcnt.
// ISA 15.18.3 opcode 98: global_load_async_to_lds_b128 vdst(LDS addr), vaddr64, off
// Generic pointers to __shared__ carry the LDS offset in their low 32 bits
// (ISA 10.2 aperture rule).
__device__ __forceinline__ void async_b128(const bf16_t* g, bf16_t* l) {
  uint32_t loff = (uint32_t)(uintptr_t)l;
  asm volatile("global_load_async_to_lds_b128 %0, %1, off"
               :
               : "v"(loff), "v"(g)
               : "memory");
}

__device__ __forceinline__ void wait_asynccnt0() {
#if __has_builtin(__builtin_amdgcn_s_wait_asynccnt)
  __builtin_amdgcn_s_wait_asynccnt(0);
#else
  asm volatile("s_wait_asynccnt 0x0" ::: "memory");
#endif
}

union FragBF {
  v16bf v;
  v8bf  h[2];
};

__device__ __forceinline__ v8f zero8() {
  v8f z;
#pragma unroll
  for (int i = 0; i < 8; ++i) z[i] = 0.0f;
  return z;
}

__device__ __forceinline__ v8f wmma_bf16(v16bf a, v16bf b, v8f c) {
  return __builtin_amdgcn_wmma_f32_16x16x32_bf16(false, a, false, b, (short)0, c,
                                                 false, false);
}

// A fragment 16(M)x32(K) from LDS [row][K] (K contiguous).
// Lanes 0-15: M=lane, K=0..7 / 16..23; lanes 16-31: M=lane-16, K=8..15 / 24..31.
__device__ __forceinline__ v16bf frag_a(const bf16_t* lds, int row0, int kk, int stride) {
  const int lane = threadIdx.x & 31;
  const bf16_t* p = lds + (row0 + (lane & 15)) * stride + kk + ((lane >> 4) << 3);
  FragBF f;
  f.h[0] = *(const v8bf*)p;
  f.h[1] = *(const v8bf*)(p + 16);
  return f.v;
}

// B fragment 32(K)x16(N) from LDS [N][K] (K contiguous).
// Lanes 0-15: N=lane, K=0..7 / 16..23; lanes 16-31: N=lane-16, K=8..15 / 24..31.
__device__ __forceinline__ v16bf frag_b(const bf16_t* lds, int row0, int kk, int stride) {
  const int lane = threadIdx.x & 31;
  const bf16_t* p = lds + (row0 + (lane & 15)) * stride + kk + ((lane >> 4) << 3);
  FragBF f;
  f.h[0] = *(const v8bf*)p;
  f.h[1] = *(const v8bf*)(p + 16);
  return f.v;
}

// ---------------------------------------------------------------------------
// Convert x / weights f32 -> bf16 (8 elements per thread)
// ---------------------------------------------------------------------------
__global__ __launch_bounds__(256) void convert_inputs(
    const float* __restrict__ x, const float* __restrict__ wqkv,
    const float* __restrict__ wp, bf16_t* __restrict__ xb,
    bf16_t* __restrict__ wq16, bf16_t* __restrict__ wp16) {
  const int j = blockIdx.x * 256 + threadIdx.x;  // 8-element chunk id
  {
    const float* s = x + (size_t)j * 8;
    v4f a = *(const v4f*)(s);
    v4f b = *(const v4f*)(s + 4);
    v8bf o;
#pragma unroll
    for (int i = 0; i < 4; ++i) {
      o[i] = (bf16_t)a[i];
      o[4 + i] = (bf16_t)b[i];
    }
    *(v8bf*)(xb + (size_t)j * 8) = o;
  }
  if (j < 98304) {  // 1536*512/8
    const float* s = wqkv + (size_t)j * 8;
    v4f a = *(const v4f*)(s);
    v4f b = *(const v4f*)(s + 4);
    v8bf o;
#pragma unroll
    for (int i = 0; i < 4; ++i) {
      o[i] = (bf16_t)a[i];
      o[4 + i] = (bf16_t)b[i];
    }
    *(v8bf*)(wq16 + (size_t)j * 8) = o;
  }
  if (j < 32768) {  // 512*512/8
    const float* s = wp + (size_t)j * 8;
    v4f a = *(const v4f*)(s);
    v4f b = *(const v4f*)(s + 4);
    v8bf o;
#pragma unroll
    for (int i = 0; i < 4; ++i) {
      o[i] = (bf16_t)a[i];
      o[4 + i] = (bf16_t)b[i];
    }
    *(v8bf*)(wp16 + (size_t)j * 8) = o;
  }
}

// ---------------------------------------------------------------------------
// Tiled GEMM: C(M,N) = A(M,K)bf16 @ W(N,K)^T + bias
//   BM=128, BN=256, BK=32, 256 threads = 8 waves (wave32)
//   wave grid 2(M) x 4(N); wave tile 64x64 = 4x4 WMMA tiles (128 acc VGPRs)
//   Double-buffered LDS; async DMA global->LDS overlapped with WMMA.
// SPLIT3: N=1536, route col/512 into q/k/v bf16 slabs; else f32 out + bias
// ---------------------------------------------------------------------------
template <bool SPLIT3>
__global__ __launch_bounds__(256) void gemm_tiled(
    const bf16_t* __restrict__ A, const bf16_t* __restrict__ W,
    const float* __restrict__ bias, bf16_t* __restrict__ outq,
    bf16_t* __restrict__ outk, bf16_t* __restrict__ outv,
    float* __restrict__ outf, int M, int N, int K) {
  constexpr int BM = 128, BN = 256, BK = 32, STR = BK + 8;
  __shared__ __align__(16) bf16_t sA[2][BM * STR];
  __shared__ __align__(16) bf16_t sB[2][BN * STR];

  const int t = threadIdx.x;
  const int lane = t & 31;
  const int wid = t >> 5;
  const int wm = wid & 1;   // M offset 64*wm
  const int wn = wid >> 1;  // N offset 64*wn
  const int mblock = blockIdx.y * BM;
  const int nblock = blockIdx.x * BN;

  v8f acc[4][4];
#pragma unroll
  for (int i = 0; i < 4; ++i)
#pragma unroll
    for (int j = 0; j < 4; ++j) acc[i][j] = zero8();

  const int arow = t >> 1;        // 0..127
  const int acol = (t & 1) << 4;  // 0 or 16
  const bf16_t* aSrc = A + (size_t)(mblock + arow) * K + acol;
  const bf16_t* bSrc = W + (size_t)(nblock + t) * K;  // one row per thread

  const int NIT = K / BK;

  auto compute = [&](int buf) {
    v16bf af[4], bfr[4];
#pragma unroll
    for (int i = 0; i < 4; ++i) af[i] = frag_a(&sA[buf][0], wm * 64 + i * 16, 0, STR);
#pragma unroll
    for (int j = 0; j < 4; ++j) bfr[j] = frag_b(&sB[buf][0], wn * 64 + j * 16, 0, STR);
#pragma unroll
    for (int i = 0; i < 4; ++i)
#pragma unroll
      for (int j = 0; j < 4; ++j) acc[i][j] = wmma_bf16(af[i], bfr[j], acc[i][j]);
  };

  auto stage = [&](int it, int buf) {
    const bf16_t* as = aSrc + it * BK;
    async_b128(as, &sA[buf][arow * STR + acol]);
    async_b128(as + 8, &sA[buf][arow * STR + acol + 8]);
    const bf16_t* bs = bSrc + it * BK;
#pragma unroll
    for (int c = 0; c < 4; ++c)
      async_b128(bs + c * 8, &sB[buf][t * STR + c * 8]);
  };

  stage(0, 0);
  for (int it = 0; it < NIT; ++it) {
    wait_asynccnt0();  // this wave's DMA for tile `it` complete
    __syncthreads();   // all waves' DMA landed; prev compute done
    if (it + 1 < NIT) stage(it + 1, (it + 1) & 1);  // overlap with compute
    compute(it & 1);
  }

  // ---- epilogue ----
#pragma unroll
  for (int i = 0; i < 4; ++i) {
#pragma unroll
    for (int j = 0; j < 4; ++j) {
      const int col = nblock + wn * 64 + j * 16 + (lane & 15);
      const int mbase = mblock + wm * 64 + i * 16 + ((lane >> 4) << 3);
      const float bv = bias[col];
      if constexpr (SPLIT3) {
        const int sel = col >> 9;
        const int cin = col & 511;
        bf16_t* dst = (sel == 0) ? outq : (sel == 1 ? outk : outv);
#pragma unroll
        for (int r = 0; r < 8; ++r)
          dst[(size_t)(mbase + r) * 512 + cin] = (bf16_t)(acc[i][j][r] + bv);
      } else {
#pragma unroll
        for (int r = 0; r < 8; ++r)
          outf[(size_t)(mbase + r) * N + col] = acc[i][j][r] + bv;
      }
    }
  }
}

// ---------------------------------------------------------------------------
// scores = Q^T K / sqrt(d) per (b,h), fused softmax over e.
// Q,K per-head slabs are contiguous (4096 x 64) bf16 (flat reinterpretation).
// ---------------------------------------------------------------------------
__global__ __launch_bounds__(256) void scores_softmax(
    const bf16_t* __restrict__ q, const bf16_t* __restrict__ k,
    bf16_t* __restrict__ watt) {
  constexpr int STR = 40;
  __shared__ __align__(16) bf16_t sQT[64 * STR];  // [d][n-chunk]
  __shared__ __align__(16) bf16_t sKT[64 * STR];  // [e][n-chunk]
  __shared__ float sS[64 * 65];

  const int head = blockIdx.x;
  const bf16_t* qs = q + (size_t)head * 4096 * 64;
  const bf16_t* ks = k + (size_t)head * 4096 * 64;

  const int t = threadIdx.x;
  const int lane = t & 31;
  const int wid = t >> 5;
  const int dt = wid & 3;         // d tile
  const int e0 = (wid >> 2) * 2;  // e tiles e0, e0+1

  v8f acc[2];
  acc[0] = zero8();
  acc[1] = zero8();

  const int nrow = t >> 3;         // 0..31
  const int dbase = (t & 7) << 3;  // 0..56

  for (int n0 = 0; n0 < 4096; n0 += 32) {
    const v8bf qv = *(const v8bf*)(qs + (size_t)(n0 + nrow) * 64 + dbase);
    const v8bf kv = *(const v8bf*)(ks + (size_t)(n0 + nrow) * 64 + dbase);
    __syncthreads();
#pragma unroll
    for (int i = 0; i < 8; ++i) {
      sQT[(dbase + i) * STR + nrow] = qv[i];
      sKT[(dbase + i) * STR + nrow] = kv[i];
    }
    __syncthreads();
    const v16bf a = frag_a(sQT, dt * 16, 0, STR);
    const v16bf b0 = frag_b(sKT, e0 * 16, 0, STR);
    const v16bf b1 = frag_b(sKT, (e0 + 1) * 16, 0, STR);
    acc[0] = wmma_bf16(a, b0, acc[0]);
    acc[1] = wmma_bf16(a, b1, acc[1]);
  }

#pragma unroll
  for (int jt = 0; jt < 2; ++jt) {
    const int e = (e0 + jt) * 16 + (lane & 15);
    const int db = dt * 16 + ((lane >> 4) << 3);
#pragma unroll
    for (int r = 0; r < 8; ++r) sS[(db + r) * 65 + e] = acc[jt][r] * 0.125f;
  }
  __syncthreads();

  if (t < 64) {
    float mx = -1e30f;
#pragma unroll
    for (int e = 0; e < 64; ++e) mx = fmaxf(mx, sS[t * 65 + e]);
    float s = 0.0f;
#pragma unroll
    for (int e = 0; e < 64; ++e) s += __expf(sS[t * 65 + e] - mx);
    const float inv = 1.0f / s;
#pragma unroll
    for (int e = 0; e < 64; ++e)
      watt[(size_t)head * 4096 + t * 64 + e] =
          (bf16_t)(__expf(sS[t * 65 + e] - mx) * inv);
  }
}

// ---------------------------------------------------------------------------
// y^T = V @ Watt^T per head: (4096 x 64) @ (64 x 64)^T
// Stored to ybuf[b, m, hh*64 + d] (lanes stripe over d -> coalesced).
// ---------------------------------------------------------------------------
__global__ __launch_bounds__(256) void attn_y(
    const bf16_t* __restrict__ v, const bf16_t* __restrict__ watt,
    bf16_t* __restrict__ y) {
  constexpr int STR = 72;
  __shared__ __align__(16) bf16_t sV[128 * STR];  // [m][e]
  __shared__ __align__(16) bf16_t sW[64 * STR];   // [d][e]

  const int mb = blockIdx.x * 128;
  const int head = blockIdx.y;
  const int bb = head >> 3, hh = head & 7;
  const bf16_t* vs = v + (size_t)head * 4096 * 64;
  const bf16_t* ws = watt + (size_t)head * 4096;

  const int t = threadIdx.x, lane = t & 31, wid = t >> 5;
  const int wm = wid & 3;   // M offset 32*wm
  const int wn = wid >> 2;  // d offset 32*wn

  const int vrow = t >> 1, vhalf = (t & 1) << 5;
  const int wrow = t >> 2, wcol = (t & 3) << 4;
  {
    const bf16_t* src = vs + (size_t)(mb + vrow) * 64 + vhalf;
    bf16_t* dst = sV + vrow * STR + vhalf;
#pragma unroll
    for (int c = 0; c < 2; ++c) async_b128(src + c * 8, dst + c * 8);
    const bf16_t* src2 = ws + wrow * 64 + wcol;
    bf16_t* dst2 = sW + wrow * STR + wcol;
#pragma unroll
    for (int c = 0; c < 2; ++c) async_b128(src2 + c * 8, dst2 + c * 8);
  }
  wait_asynccnt0();
  __syncthreads();

  v8f acc[2][2];
#pragma unroll
  for (int i = 0; i < 2; ++i)
#pragma unroll
    for (int j = 0; j < 2; ++j) acc[i][j] = zero8();

#pragma unroll
  for (int kk = 0; kk < 64; kk += 32) {
    v16bf af[2], bfr[2];
#pragma unroll
    for (int i = 0; i < 2; ++i) af[i] = frag_a(sV, wm * 32 + i * 16, kk, STR);
#pragma unroll
    for (int j = 0; j < 2; ++j) bfr[j] = frag_b(sW, wn * 32 + j * 16, kk, STR);
#pragma unroll
    for (int i = 0; i < 2; ++i)
#pragma unroll
      for (int j = 0; j < 2; ++j) acc[i][j] = wmma_bf16(af[i], bfr[j], acc[i][j]);
  }

#pragma unroll
  for (int i = 0; i < 2; ++i) {
#pragma unroll
    for (int j = 0; j < 2; ++j) {
      const int d = wn * 32 + j * 16 + (lane & 15);
      const int mbase = mb + wm * 32 + i * 16 + ((lane >> 4) << 3);
#pragma unroll
      for (int r = 0; r < 8; ++r)
        y[((size_t)bb * 4096 + (mbase + r)) * 512 + hh * 64 + d] =
            (bf16_t)acc[i][j][r];
    }
  }
}

// ---------------------------------------------------------------------------
extern "C" void kernel_launch(void* const* d_in, const int* in_sizes, int n_in,
                              void* d_out, int out_size, void* d_ws,
                              size_t ws_size, hipStream_t stream) {
  const float* x      = (const float*)d_in[0];
  const float* wqkv_w = (const float*)d_in[1];
  const float* wqkv_b = (const float*)d_in[2];
  const float* wp_w   = (const float*)d_in[3];
  const float* wp_b   = (const float*)d_in[4];
  float* out = (float*)d_out;

  char* p = (char*)d_ws;
  const size_t SLAB = (size_t)8 * 4096 * 512 * sizeof(bf16_t);  // 33.5 MB
  bf16_t* xb   = (bf16_t*)p; p += SLAB;
  bf16_t* wq16 = (bf16_t*)p; p += 1536 * 512 * sizeof(bf16_t);
  bf16_t* wp16 = (bf16_t*)p; p += 512 * 512 * sizeof(bf16_t);
  bf16_t* qb   = (bf16_t*)p; p += SLAB;
  bf16_t* kb   = (bf16_t*)p; p += SLAB;
  bf16_t* vb   = (bf16_t*)p; p += SLAB;
  bf16_t* watt = (bf16_t*)p; p += 64 * 64 * 64 * sizeof(bf16_t);
  bf16_t* yb   = (bf16_t*)p;

  // x (16.78M elems) / wqkv / wp -> bf16
  convert_inputs<<<8192, 256, 0, stream>>>(x, wqkv_w, wp_w, xb, wq16, wp16);

  // QKV projection: (32768 x 512) @ (1536 x 512)^T -> q/k/v bf16 slabs
  gemm_tiled<true><<<dim3(6, 256), 256, 0, stream>>>(
      xb, wq16, wqkv_b, qb, kb, vb, nullptr, 32768, 1536, 512);

  // channel attention scores + softmax (64 heads)
  scores_softmax<<<64, 256, 0, stream>>>(qb, kb, watt);

  // y^T = V @ Watt^T, written as (b, n, c) bf16
  attn_y<<<dim3(32, 64), 256, 0, stream>>>(vb, watt, yb);

  // output projection: (32768 x 512) @ (512 x 512)^T + bias -> f32 out
  gemm_tiled<false><<<dim3(2, 256), 256, 0, stream>>>(
      yb, wp16, wp_b, nullptr, nullptr, nullptr, out, 32768, 512, 512);
}
